// DCIM_55250459295990
// MI455X (gfx1250) — matrix operand order repaired and checked by reading, not compiled, not run on hardware
//
#include <hip/hip_runtime.h>
#include <hip/hip_bf16.h>

typedef unsigned short u16;
typedef __attribute__((ext_vector_type(16))) __bf16 bf16x16;
typedef __attribute__((ext_vector_type(8)))  float  floatx8;

union Frag {
    bf16x16 v;
    uint4   q[2];
};

__device__ __forceinline__ u16 f2bf(float f) {
    unsigned u = __float_as_uint(f);
    // round-to-nearest-even bf16
    return (u16)((u + 0x7FFFu + ((u >> 16) & 1u)) >> 16);
}

// ---------------------------------------------------------------------------
// B=16, C=512, H=W=64 (HW=4096), K=4, D=32 -> stacked hidden = K*D = 128
// ---------------------------------------------------------------------------
#define CC    512
#define HW    4096
#define KD    128
#define PXT   64            // pixel tile per workgroup
#define XS_S  136           // LDS stride of x K-panel (128 + 8 pad), bank-friendly
#define HID_S 136           // LDS stride of hid (128 + 8 pad)
#define WL_S  136           // LDS stride of W1 K-panel

#if __has_builtin(__builtin_amdgcn_global_load_async_to_lds_b128)
#define USE_ASYNC_LDS 1
// b128 async builtin is typed over GCC int4 vectors with AS-qualified pointers:
// (global src, lds dst, imm offset, imm cpol)
typedef int v4i __attribute__((vector_size(16)));
typedef __attribute__((address_space(1))) v4i* gptr_b128;
typedef __attribute__((address_space(3))) v4i* lptr_b128;
#else
#define USE_ASYNC_LDS 0
#endif

// ------------------------- prep kernels ------------------------------------

// global average pool: one block per (b,c)
__global__ __launch_bounds__(256) void pool_kernel(const float* __restrict__ x,
                                                   float* __restrict__ y) {
    __shared__ float red[256];
    size_t bc = blockIdx.x;
    const float* p = x + bc * (size_t)HW;
    float s = 0.f;
    for (int i = threadIdx.x; i < HW; i += 256) s += p[i];
    red[threadIdx.x] = s;
    __syncthreads();
    for (int off = 128; off > 0; off >>= 1) {
        if (threadIdx.x < off) red[threadIdx.x] += red[threadIdx.x + off];
        __syncthreads();
    }
    if (threadIdx.x == 0) y[bc] = red[0] * (1.0f / (float)HW);
}

// routing logits + softmax: one block of 64 threads (16 batches x 4 experts)
__global__ void route_kernel(const float* __restrict__ y,
                             const float* __restrict__ fcw,
                             const float* __restrict__ fcb,
                             float* __restrict__ wts) {
    __shared__ float lg[64];
    int t = threadIdx.x;
    int b = t >> 2, k = t & 3;
    float s = fcb[k];
    const float* yr = y + b * CC;
    const float* wr = fcw + k * CC;
    for (int c = 0; c < CC; ++c) s += yr[c] * wr[c];
    lg[t] = s;
    __syncthreads();
    if (k == 0) {
        float m = lg[t];
        for (int j = 1; j < 4; ++j) m = fmaxf(m, lg[t + j]);
        float e[4], sum = 0.f;
        for (int j = 0; j < 4; ++j) { e[j] = __expf(lg[t + j] - m); sum += e[j]; }
        for (int j = 0; j < 4; ++j) wts[b * 4 + j] = e[j] / sum;
    }
}

// convert w1 [K,D,C] (already [128][512] row-major) to bf16
__global__ __launch_bounds__(256) void prep_w1_kernel(const float* __restrict__ w1,
                                                      u16* __restrict__ w1bf) {
    int idx = blockIdx.x * 256 + threadIdx.x;   // 65536 total
    w1bf[idx] = f2bf(w1[idx]);
}

// W2w[b][c][kd] = wts[b,k] * w2[k][c][d], kd = k*32 + d  (bf16)
__global__ __launch_bounds__(256) void prep_w2_kernel(const float* __restrict__ w2,
                                                      const float* __restrict__ wts,
                                                      u16* __restrict__ W2w) {
    int idx = blockIdx.x * 256 + threadIdx.x;   // 16*512*128 = 1048576 total
    int b  = idx >> 16;
    int c  = (idx >> 7) & 511;
    int kd = idx & 127;
    int ke = kd >> 5;            // expert index
    int dd = kd & 31;            // within-expert hidden channel
    float s = wts[b * 4 + ke] * w2[((size_t)ke * CC + c) * 32 + dd];
    W2w[idx] = f2bf(s);
}

// bias2w[b][c] = sum_k wts[b,k]*b2[k,c]
__global__ __launch_bounds__(256) void prep_b2_kernel(const float* __restrict__ b2,
                                                      const float* __restrict__ wts,
                                                      float* __restrict__ bias2w) {
    int idx = blockIdx.x * 256 + threadIdx.x;   // 8192 total
    int b = idx >> 9, c = idx & 511;
    float s = 0.f;
#pragma unroll
    for (int k = 0; k < 4; ++k) s += wts[b * 4 + k] * b2[k * CC + c];
    bias2w[idx] = s;
}

// ------------------------- fused main kernel -------------------------------
// grid (HW/PXT, B), block 256 (8 waves of 32)
__global__ __launch_bounds__(256) void moe_se_kernel(const float* __restrict__ x,
                                                     const u16* __restrict__ w1bf,
                                                     const float* __restrict__ b1,
                                                     const u16* __restrict__ W2w,
                                                     const float* __restrict__ bias2w,
                                                     float* __restrict__ out) {
    __shared__ u16 xs[PXT * XS_S];    // x K-panel, transposed [px][128 C], bf16
    __shared__ u16 hidl[PXT * HID_S]; // hidden activations [px][128 kd], bf16
#if USE_ASYNC_LDS
    __shared__ u16 wl1[KD * WL_S];    // W1 K-panel [128 kd][128 C], bf16
#endif

    const int b   = blockIdx.y;
    const int px0 = blockIdx.x * PXT;
    const float* xb = x + (size_t)b * CC * HW;

    const int t  = threadIdx.x;
    const int wv = t >> 5;           // wave 0..7
    const int ln = t & 31;
    const int lh = ln >> 4;          // lane half
    const int ll = ln & 15;

    // ---------------- GEMM1: hid = relu(W1[128x512] @ Xt + b1) -------------
    floatx8 acc[4];
#pragma unroll
    for (int n = 0; n < 4; ++n) acc[n] = (floatx8){0.f,0.f,0.f,0.f,0.f,0.f,0.f,0.f};

#if !USE_ASYNC_LDS
    const u16* arow = w1bf + (size_t)(wv * 16 + ll) * CC;   // A row for this lane
#endif

    for (int kp = 0; kp < 4; ++kp) {                        // K panels of 128
#if USE_ASYNC_LDS
        // async-stage W1 panel [128 rows][128 K] (bf16, raw copy) into LDS;
        // overlaps with the VALU fp32->bf16 x staging below.
#pragma unroll
        for (int i = 0; i < 8; ++i) {
            int e   = t + i * 256;            // 2048 x 16B chunks
            int row = e >> 4;
            int cs  = (e & 15) * 8;
            __builtin_amdgcn_global_load_async_to_lds_b128(
                (gptr_b128)(w1bf + (size_t)row * CC + kp * 128 + cs),
                (lptr_b128)&wl1[row * WL_S + cs], 0, 0);
        }
#endif
        // stage x panel -> bf16 transposed into LDS
        for (int e = t; e < PXT * 128; e += 256) {
            int c = e >> 6;          // 0..127 within panel
            int p = e & 63;          // pixel within tile
            xs[p * XS_S + c] = f2bf(xb[(size_t)(kp * 128 + c) * HW + px0 + p]);
        }
#if USE_ASYNC_LDS
        asm volatile("s_wait_asynccnt 0x0" ::: "memory");
#endif
        __syncthreads();

#pragma unroll
        for (int kk = 0; kk < 4; ++kk) {                    // 4 x K=32 steps
            Frag a;
#if USE_ASYNC_LDS
            // A 16x32 bf16 layout: lane half lh holds K = 8*lh+{0..7} and +16
            const u16* ap = &wl1[(wv * 16 + ll) * WL_S + kk * 32 + 8 * lh];
            a.q[0] = *(const uint4*)ap;
            a.q[1] = *(const uint4*)(ap + 16);
#else
            const int kg = kp * 128 + kk * 32;
            a.q[0] = *(const uint4*)(arow + kg + 8 * lh);
            a.q[1] = *(const uint4*)(arow + kg + 8 * lh + 16);
#endif
            // load all 4 B fragments first, then issue the WMMA chain
            Frag bb[4];
#pragma unroll
            for (int n = 0; n < 4; ++n) {
                // B 32x16 bf16 layout: lane half lh holds K = 8*lh+{0..7} and +16
                const u16* bp = &xs[(n * 16 + ll) * XS_S + kk * 32 + 8 * lh];
                bb[n].q[0] = *(const uint4*)bp;
                bb[n].q[1] = *(const uint4*)(bp + 16);
            }
#pragma unroll
            for (int n = 0; n < 4; ++n)
                acc[n] = __builtin_amdgcn_wmma_f32_16x16x32_bf16(
                    false, a.v, false, bb[n].v, (short)0, acc[n], false, false);
        }
        __syncthreads();
    }

    // epilogue: + b1, relu, pack bf16 -> hidl  (D row M = r + 8*lh)
    float bias1[8];
#pragma unroll
    for (int r = 0; r < 8; ++r) bias1[r] = b1[wv * 16 + 8 * lh + r];
#pragma unroll
    for (int n = 0; n < 4; ++n) {
        unsigned pr[4];
#pragma unroll
        for (int h = 0; h < 4; ++h) {
            float v0 = acc[n][2 * h]     + bias1[2 * h];     v0 = v0 > 0.f ? v0 : 0.f;
            float v1 = acc[n][2 * h + 1] + bias1[2 * h + 1]; v1 = v1 > 0.f ? v1 : 0.f;
            pr[h] = (unsigned)f2bf(v0) | ((unsigned)f2bf(v1) << 16);
        }
        uint4 pk; pk.x = pr[0]; pk.y = pr[1]; pk.z = pr[2]; pk.w = pr[3];
        int p = n * 16 + ll;
        *(uint4*)&hidl[p * HID_S + wv * 16 + 8 * lh] = pk;
    }
    __syncthreads();

    // ---------------- GEMM2: out = W2w[b][512x128] @ hid ------------------
    const int nt = wv & 3;            // N tile (16 px) owned by this wave
    const int mh = wv >> 2;           // M half (rows 0..255 or 256..511)

    Frag bh[4];                       // preload all K fragments of our N tile
#pragma unroll
    for (int kk = 0; kk < 4; ++kk) {
        const u16* bp = &hidl[(nt * 16 + ll) * HID_S + kk * 32 + 8 * lh];
        bh[kk].q[0] = *(const uint4*)bp;
        bh[kk].q[1] = *(const uint4*)(bp + 16);
    }

    const u16*   w2b  = W2w + (size_t)b * CC * KD;
    const float* b2b  = bias2w + b * CC;
    float*       outb = out + (size_t)b * CC * HW;
    const int    px   = px0 + nt * 16 + ll;

    for (int m = mh * 16; m < mh * 16 + 16; ++m) {          // 16 M tiles
        const u16* ar = w2b + (size_t)(m * 16 + ll) * KD;
        __builtin_prefetch(ar + 16 * KD, 0, 0);             // next tile's rows
        // load all 4 A fragments first, then issue the WMMA chain
        Frag a4[4];
#pragma unroll
        for (int kk = 0; kk < 4; ++kk) {
            a4[kk].q[0] = *(const uint4*)(ar + kk * 32 + 8 * lh);
            a4[kk].q[1] = *(const uint4*)(ar + kk * 32 + 8 * lh + 16);
        }
        floatx8 c = (floatx8){0.f,0.f,0.f,0.f,0.f,0.f,0.f,0.f};
#pragma unroll
        for (int kk = 0; kk < 4; ++kk)
            c = __builtin_amdgcn_wmma_f32_16x16x32_bf16(
                false, a4[kk].v, false, bh[kk].v, (short)0, c, false, false);
#pragma unroll
        for (int r = 0; r < 8; ++r) {
            int ch = m * 16 + 8 * lh + r;
            float v = c[r] + b2b[ch];
            float attn = 1.f / (1.f + __expf(-v));
            outb[(size_t)ch * HW + px] = xb[(size_t)ch * HW + px] * attn;
        }
    }
}

// ---------------------------------------------------------------------------

extern "C" void kernel_launch(void* const* d_in, const int* in_sizes, int n_in,
                              void* d_out, int out_size, void* d_ws, size_t ws_size,
                              hipStream_t stream) {
    (void)in_sizes; (void)n_in; (void)out_size; (void)ws_size;
    const float* x    = (const float*)d_in[0];   // [16,512,64,64]
    const float* fc_w = (const float*)d_in[1];   // [4,512]
    const float* fc_b = (const float*)d_in[2];   // [4]
    const float* w1   = (const float*)d_in[3];   // [4,32,512]
    const float* b1   = (const float*)d_in[4];   // [4,32] == [128]
    const float* w2   = (const float*)d_in[5];   // [4,512,32]
    const float* b2   = (const float*)d_in[6];   // [4,512]
    float* out = (float*)d_out;

    // workspace layout
    float* y      = (float*)d_ws;                 // 16*512
    float* wts    = y + 16 * CC;                  // 16*4 (pad to 64)
    float* bias2w = wts + 64;                     // 16*512
    u16*   w1bf   = (u16*)(bias2w + 16 * CC);     // 128*512 bf16
    u16*   W2w    = w1bf + KD * CC;               // 16*512*128 bf16

    pool_kernel   <<<16 * CC, 256, 0, stream>>>(x, y);
    route_kernel  <<<1, 64, 0, stream>>>(y, fc_w, fc_b, wts);
    prep_w1_kernel<<<(KD * CC) / 256, 256, 0, stream>>>(w1, w1bf);
    prep_w2_kernel<<<(16 * CC * KD) / 256, 256, 0, stream>>>(w2, wts, W2w);
    prep_b2_kernel<<<(16 * CC) / 256, 256, 0, stream>>>(b2, wts, bias2w);

    dim3 grid(HW / PXT, 16);
    moe_se_kernel<<<grid, 256, 0, stream>>>(x, w1bf, b1, W2w, bias2w, out);
}
